// Decoder_50775103373425
// MI455X (gfx1250) — matrix operand order repaired and checked by reading, not compiled, not run on hardware
//
#include <hip/hip_runtime.h>

// ---------------------------------------------------------------------------
// Autoregressive LSTM decoder for MI455X (gfx1250, wave32, WMMA).
//
// Formulation per 16-batch tile (one wave owns a tile for all 25 steps):
//   gates(16g x 16b) = W_tile(16g x K) x xh^T(K x 16b)   via v_wmma_f32_16x16x32_f16
// Weights are the A operand (pre-swizzled per-lane images in LDS, loaded once
// per block), activations are the B operand (per-wave LDS staging buffer,
// row-major f16, row stride 144B). D layout (lane = batch col) matches the B
// layout's batch-per-lane mapping, so h feeds the next step with only
// f32->f16 packing + one ds_store_b128 per 16x16 tile (no lane transpose).
// Bias is folded in as a constant-1 feature at K=48 (K padded 48 -> 64).
// Projection bias rides in the WMMA C operand.
// ---------------------------------------------------------------------------

typedef __attribute__((ext_vector_type(16))) _Float16 v16h;
typedef __attribute__((ext_vector_type(8)))  _Float16 v8h;
typedef __attribute__((ext_vector_type(8)))  float    v8f;

#define SEQ_LEN 25
#define BATCH   262144

__device__ __forceinline__ unsigned pack2f16(float a, float b) {
  union { _Float16 h[2]; unsigned u; } x;
  x.h[0] = (_Float16)a; x.h[1] = (_Float16)b;
  return x.u;
}

#if __has_builtin(__builtin_amdgcn_tanhf)
__device__ __forceinline__ float t_tanh(float x) { return __builtin_amdgcn_tanhf(x); }
#else
__device__ __forceinline__ float t_tanh(float x) {
  float e = __expf(2.0f * x);
  return 1.0f - 2.0f / (e + 1.0f);
}
#endif
__device__ __forceinline__ float t_sig(float x) { return 0.5f * t_tanh(0.5f * x) + 0.5f; }

__device__ __forceinline__ v16h ld_a16(const unsigned* p) {  // 32B-aligned
  return *(const v16h*)p;
}
__device__ __forceinline__ v16h ld_b16(const unsigned* p) {  // 16B-aligned
  v8h lo = *(const v8h*)p;
  v8h hh = *(const v8h*)(p + 4);
  return __builtin_shufflevector(lo, hh, 0,1,2,3,4,5,6,7,8,9,10,11,12,13,14,15);
}

// Two chained K=32 WMMAs covering the padded K=64 feature axis.
__device__ __forceinline__ v8f wmma_pair(const unsigned* wl, int gt, int lane,
                                         v16h b0, v16h b1) {
  v16h a0 = ld_a16(wl + ((gt * 2 + 0) * 256 + lane * 8));
  v16h a1 = ld_a16(wl + ((gt * 2 + 1) * 256 + lane * 8));
  v8f z = {};
  v8f c = __builtin_amdgcn_wmma_f32_16x16x32_f16(false, a0, false, b0,
                                                 (short)0, z, false, false);
  c = __builtin_amdgcn_wmma_f32_16x16x32_f16(false, a1, false, b1,
                                             (short)0, c, false, false);
  return c;
}

__global__ __launch_bounds__(256)
void lstm_decoder_kernel(const float* __restrict__ x0,
                         const float* __restrict__ h0,
                         const float* __restrict__ c0,
                         const float* __restrict__ Wih,   // [128,16]
                         const float* __restrict__ Whh,   // [128,32]
                         const float* __restrict__ bih,   // [128]
                         const float* __restrict__ bhh,   // [128]
                         const float* __restrict__ Wout,  // [16,32]
                         const float* __restrict__ bout,  // [16]
                         float* __restrict__ out)         // [B,25,16]
{
  // 16 gate A-tiles (8 gate groups x 2 K-blocks) + 1 projection A-tile.
  // Each tile = per-lane image: 32 lanes x 8 dwords (16 f16 of one A row-half).
  __shared__ __align__(32) unsigned wlds[17 * 256];        // 17408 B
  // Per-wave activation staging: 16 rows x 144B (64 f16 feats: x16|h32|1|0pad)
  __shared__ __align__(16) unsigned abuf[8 * 16 * 36];     // 18432 B

  const int tid  = threadIdx.x;
  const int lane = tid & 31;
  const int wid  = tid >> 5;
  const int n    = lane & 15;   // batch column inside tile
  const int hi   = lane >> 4;   // lane half

  // ---- Stage weights into LDS as pre-swizzled A-operand images ----
  for (int idx = tid; idx < 17 * 256; idx += 256) {
    int tile = idx >> 8;
    int rem  = idx & 255;
    int l    = rem >> 3;        // lane image 0..31
    int v    = rem & 7;         // dword slot (VGPR index)
    int mrow = l & 15;
    int lh   = l >> 4;
    float f0, f1;
    if (tile < 16) {
      int gt   = tile >> 1, kb = tile & 1;
      int gate = gt * 16 + mrow;
      int k0   = kb * 32 + lh * 16 + v * 2;
      float w[2];
#pragma unroll
      for (int j = 0; j < 2; ++j) {
        int k = k0 + j;
        float val;
        if (k < 16)       val = Wih[gate * 16 + k];
        else if (k < 48)  val = Whh[gate * 32 + (k - 16)];
        else if (k == 48) val = bih[gate] + bhh[gate];   // bias feature
        else              val = 0.0f;
        w[j] = val;
      }
      f0 = w[0]; f1 = w[1];
    } else {                    // projection tile: W_out, K = 32 exactly
      int p  = mrow;
      int k0 = lh * 16 + v * 2;
      f0 = Wout[p * 32 + k0];
      f1 = Wout[p * 32 + k0 + 1];
    }
    wlds[idx] = pack2f16(f0, f1);
  }

  // ---- Per-wave activation buffer init ----
  unsigned* ab = abuf + wid * (16 * 36);
  const long tileBase = ((long)blockIdx.x * 8 + wid) * 16;
  const long brow     = tileBase + n;

  { // x feats 0..15: this lane covers feats hi*8 .. hi*8+7
    const float* xr = x0 + brow * 16 + hi * 8;
    uint4 w = make_uint4(pack2f16(xr[0], xr[1]), pack2f16(xr[2], xr[3]),
                         pack2f16(xr[4], xr[5]), pack2f16(xr[6], xr[7]));
    *(uint4*)(ab + n * 36 + hi * 4) = w;
  }
  { // h feats 16..47: this lane covers h[16hi .. 16hi+15] -> dwords 8+8hi..
    const float* hr = h0 + brow * 32 + hi * 16;
    uint4 w1 = make_uint4(pack2f16(hr[0], hr[1]),   pack2f16(hr[2], hr[3]),
                          pack2f16(hr[4], hr[5]),   pack2f16(hr[6], hr[7]));
    uint4 w2 = make_uint4(pack2f16(hr[8], hr[9]),   pack2f16(hr[10], hr[11]),
                          pack2f16(hr[12], hr[13]), pack2f16(hr[14], hr[15]));
    *(uint4*)(ab + n * 36 + 8 + hi * 8)     = w1;
    *(uint4*)(ab + n * 36 + 8 + hi * 8 + 4) = w2;
  }
  if (hi == 0) {  // feat 48 = 1.0 (bias feature), feats 49..55 = 0
    *(uint4*)(ab + n * 36 + 24) = make_uint4(0x00003C00u, 0u, 0u, 0u);
  } else {        // feats 56..63 = 0
    *(uint4*)(ab + n * 36 + 28) = make_uint4(0u, 0u, 0u, 0u);
  }

  // Cell state in registers, D-layout: cst[t][r] = c[batch n][j = 16t + r + 8hi]
  float cst[2][8];
#pragma unroll
  for (int t = 0; t < 2; ++t)
#pragma unroll
    for (int r = 0; r < 8; ++r)
      cst[t][r] = c0[brow * 32 + 16 * t + r + 8 * hi];

  // Projection bias as WMMA C operand: cbout[r] = b_out[r + 8hi]
  v8f cbout;
#pragma unroll
  for (int r = 0; r < 8; ++r) cbout[r] = bout[r + 8 * hi];

  __syncthreads();

  // ---- Sequence loop (serial dependency; all LDS traffic is wave-private,
  //      LDS ops from one wave are in-order -> no barriers needed) ----
#pragma unroll 1
  for (int s = 0; s < SEQ_LEN; ++s) {
    // B operands: lane (n, hi) reads feats [16hi, 16hi+16) and [32+16hi, ...)
    v16h b0 = ld_b16(ab + n * 36 + 8 * hi);        // feats 0..31  (x | h0..15)
    v16h b1 = ld_b16(ab + n * 36 + 16 + 8 * hi);   // feats 32..63 (h16..31|1|0)

#pragma unroll
    for (int t = 0; t < 2; ++t) {
      v8f gi = wmma_pair(wlds, 0 + t, lane, b0, b1);  // i gates 16t..16t+15
      v8f gf = wmma_pair(wlds, 2 + t, lane, b0, b1);  // f
      v8f gg = wmma_pair(wlds, 4 + t, lane, b0, b1);  // g
      v8f go = wmma_pair(wlds, 6 + t, lane, b0, b1);  // o
      float hv[8];
#pragma unroll
      for (int r = 0; r < 8; ++r) {
        float iv = t_sig(gi[r]);
        float fv = t_sig(gf[r]);
        float gv = t_tanh(gg[r]);
        float ov = t_sig(go[r]);
        float cc = fv * cst[t][r] + iv * gv;
        cst[t][r] = cc;
        hv[r] = ov * t_tanh(cc);
      }
      // h[j = 16t + 8hi + 0..7] -> abuf feats 16+16t+8hi -> dwords 8+8t+4hi
      uint4 hp = make_uint4(pack2f16(hv[0], hv[1]), pack2f16(hv[2], hv[3]),
                            pack2f16(hv[4], hv[5]), pack2f16(hv[6], hv[7]));
      *(uint4*)(ab + n * 36 + 8 + 8 * t + 4 * hi) = hp;
    }

    // Projection: x' = W_out x h^T + b_out  (K = 32 = h feats 16..47 in abuf)
    v16h pb = ld_b16(ab + n * 36 + 8 + 8 * hi);
    v16h pa = ld_a16(wlds + 16 * 256 + lane * 8);
    v8f  px = __builtin_amdgcn_wmma_f32_16x16x32_f16(false, pa, false, pb,
                                                     (short)0, cbout, false, false);

    // out[b][s][p]: lane stores p = 8hi + 0..7 contiguously (2x b128)
    long ob = (brow * SEQ_LEN + s) * 16 + 8 * hi;
    *(float4*)(out + ob)     = make_float4(px[0], px[1], px[2], px[3]);
    *(float4*)(out + ob + 4) = make_float4(px[4], px[5], px[6], px[7]);

    // Feed x' back as next step's feats 0..15 (this lane: feats 8hi..8hi+7)
    uint4 xp = make_uint4(pack2f16(px[0], px[1]), pack2f16(px[2], px[3]),
                          pack2f16(px[4], px[5]), pack2f16(px[6], px[7]));
    *(uint4*)(ab + n * 36 + hi * 4) = xp;
  }
}

extern "C" void kernel_launch(void* const* d_in, const int* in_sizes, int n_in,
                              void* d_out, int out_size, void* d_ws, size_t ws_size,
                              hipStream_t stream) {
  const float* x0   = (const float*)d_in[0];
  const float* h0   = (const float*)d_in[1];
  const float* c0   = (const float*)d_in[2];
  const float* Wih  = (const float*)d_in[3];
  const float* Whh  = (const float*)d_in[4];
  const float* bih  = (const float*)d_in[5];
  const float* bhh  = (const float*)d_in[6];
  const float* Wout = (const float*)d_in[7];
  const float* bout = (const float*)d_in[8];
  float* out = (float*)d_out;

  dim3 grid(BATCH / 128);   // 16 batch rows per wave, 8 waves per block
  dim3 block(256);
  hipLaunchKernelGGL(lstm_decoder_kernel, grid, block, 0, stream,
                     x0, h0, c0, Wih, Whh, bih, bhh, Wout, bout, out);
}
